// Attention_57561151701596
// MI455X (gfx1250) — matrix operand order repaired and checked by reading, not compiled, not run on hardware
//
#include <hip/hip_runtime.h>
#include <hip/hip_bf16.h>

// ---------------------------------------------------------------------------
// CDNA5 (gfx1250) wave32 WMMA attention block.
// D = A(16x32 bf16) x B(32x16 bf16) + C(16x16 f32) via v_wmma_f32_16x16x32_bf16
// Async global->LDS staging (ASYNCcnt) + DPP row_xmask softmax reductions.
// ---------------------------------------------------------------------------

typedef __attribute__((ext_vector_type(16))) __bf16 v16bf;
typedef __attribute__((ext_vector_type(8)))  __bf16 v8bf;
typedef __attribute__((ext_vector_type(8)))  float  v8f;

__device__ __forceinline__ __bf16 f2bf(float f) {
    unsigned u = __builtin_bit_cast(unsigned, f);
    u = (u + 0x7FFFu + ((u >> 16) & 1u)) >> 16;          // round-to-nearest-even
    return __builtin_bit_cast(__bf16, (unsigned short)u);
}

__device__ __forceinline__ v16bf cat16(v8bf lo, v8bf hi) {
    return __builtin_shufflevector(lo, hi, 0,1,2,3,4,5,6,7,8,9,10,11,12,13,14,15);
}

__device__ __forceinline__ v8f wmma_bf16f32(v16bf a, v16bf b, v8f c) {
    // 8 args: (neg_a, A, neg_b, B, c_mod, C, reuse_a, reuse_b)
    return __builtin_amdgcn_wmma_f32_16x16x32_bf16(false, a, false, b, (short)0, c, false, false);
}

// DPP16 ROW_XMASK cross-lane exchange within a 16-lane row (VALU-only).
#define DPP_XOR(res, x, mask)                                                   \
    do {                                                                        \
        int _i = __builtin_bit_cast(int, (x));                                  \
        int _j = __builtin_amdgcn_update_dpp(_i, _i, 0x160 + (mask), 0xF, 0xF, true); \
        (res) = __builtin_bit_cast(float, _j);                                  \
    } while (0)

__device__ __forceinline__ float rowmax16(float x) {
    float t;
    DPP_XOR(t, x, 1); x = fmaxf(x, t);
    DPP_XOR(t, x, 2); x = fmaxf(x, t);
    DPP_XOR(t, x, 4); x = fmaxf(x, t);
    DPP_XOR(t, x, 8); x = fmaxf(x, t);
    return x;
}

__device__ __forceinline__ float rowsum16(float x) {
    float t;
    DPP_XOR(t, x, 1); x += t;
    DPP_XOR(t, x, 2); x += t;
    DPP_XOR(t, x, 4); x += t;
    DPP_XOR(t, x, 8); x += t;
    return x;
}

// Async global -> LDS 16-byte copy (no VGPR data path; tracked by ASYNCcnt).
__device__ __forceinline__ void async_ld_b128(void* lds_ptr, const void* gptr) {
    unsigned loff = (unsigned)(uintptr_t)lds_ptr;  // addrspace(3) byte offset
    asm volatile("global_load_async_to_lds_b128 %0, %1, off"
                 :: "v"(loff), "v"(gptr)
                 : "memory");
}

__device__ __forceinline__ void wait_asynccnt0() {
    asm volatile("s_wait_asynccnt 0x0" ::: "memory");
}

// ---------------------------------------------------------------------------
// fp32 -> bf16 elementwise conversion (weights)
// ---------------------------------------------------------------------------
__global__ void cvt_bf16_kernel(const float* __restrict__ in, __bf16* __restrict__ out, int n) {
    int i = blockIdx.x * blockDim.x + threadIdx.x;
    int stride = gridDim.x * blockDim.x;
    for (; i < n; i += stride) out[i] = f2bf(in[i]);
}

// ---------------------------------------------------------------------------
// LayerNorm over last dim C, fp32 in -> bf16 out. One block (128 thr) per row.
// ---------------------------------------------------------------------------
__global__ __launch_bounds__(128) void ln_bf16_kernel(
    const float* __restrict__ x, const float* __restrict__ g,
    const float* __restrict__ bta, __bf16* __restrict__ out, int C)
{
    int row = blockIdx.x;
    const float* xr = x + (size_t)row * C;
    float s = 0.f, sq = 0.f;
    for (int c = threadIdx.x; c < C; c += 128) { float v = xr[c]; s += v; sq += v * v; }
    // 16-lane butterfly via DPP, then one cross-row hop via shfl
    s = rowsum16(s); sq = rowsum16(sq);
    s += __shfl_xor(s, 16, 32); sq += __shfl_xor(sq, 16, 32);
    __shared__ float red[8];
    int w = threadIdx.x >> 5, l = threadIdx.x & 31;
    if (l == 0) { red[w] = s; red[4 + w] = sq; }
    __syncthreads();
    s  = red[0] + red[1] + red[2] + red[3];
    sq = red[4] + red[5] + red[6] + red[7];
    float mu  = s / (float)C;
    float var = sq / (float)C - mu * mu;
    float inv = rsqrtf(var + 1e-5f);
    for (int c = threadIdx.x; c < C; c += 128)
        out[(size_t)row * C + c] = f2bf((xr[c] - mu) * inv * g[c] + bta[c]);
}

// ---------------------------------------------------------------------------
// Tiled bf16 GEMM: C[M,N] = A[M,K] * B[K,N]. 128x128 macro tile, 4 waves,
// 64x64 wave tile, K-step 32. A tile staged via async global->LDS; B tile
// transposed through VGPRs. Padded LDS strides for bank safety.
// ---------------------------------------------------------------------------
#define GEMM_LDA 40   // 32 K + 8 pad, bf16 elems (80B row, 16B aligned)
#define GEMM_LDB 40

template <bool OUT_BF16, bool HAS_BIAS>
__global__ __launch_bounds__(128) void gemm_bf16_kernel(
    const __bf16* __restrict__ A, const __bf16* __restrict__ Bm,
    const float* __restrict__ bias, void* __restrict__ Cout,
    int M, int Nn, int K)
{
    __shared__ __bf16 ldsA[128 * GEMM_LDA];
    __shared__ __bf16 ldsB[128 * GEMM_LDB];

    const int tid   = threadIdx.x;
    const int lane  = tid & 31;
    const int lmod  = lane & 15;
    const int khalf = lane >> 4;
    const int wave  = tid >> 5;
    const int waveM = wave >> 1;   // 0..1
    const int waveN = wave & 1;    // 0..1

    v8f acc[4][4] = {};

    const __bf16* srcA_row = A + (size_t)(blockIdx.y * 128 + tid) * K;
    const int klB = tid >> 2;
    const int ncB = (tid & 3) * 32;

    for (int kk = 0; kk < K; kk += 32) {
        __syncthreads();
        { // A tile: 128 rows x 32 k -> async DMA straight into LDS
            const __bf16* src = srcA_row + kk;
            __bf16* dst = &ldsA[tid * GEMM_LDA];
#pragma unroll
            for (int c = 0; c < 4; ++c)
                async_ld_b128(dst + c * 8, src + c * 8);
            if (kk + 32 < K) __builtin_prefetch(src + 32, 0, 1);
        }
        { // B tile: 32 k x 128 n, transpose into ldsB[n][k] via VGPRs
            const __bf16* src = Bm + (size_t)(kk + klB) * Nn + blockIdx.x * 128 + ncB;
#pragma unroll
            for (int c = 0; c < 4; ++c) {
                v8bf vv = *(const v8bf*)(src + c * 8);
#pragma unroll
                for (int e = 0; e < 8; ++e)
                    ldsB[(ncB + c * 8 + e) * GEMM_LDB + klB] = vv[e];
            }
            if (kk + 32 < K)
                __builtin_prefetch(Bm + (size_t)(kk + 32 + klB) * Nn + blockIdx.x * 128 + ncB, 0, 1);
        }
        wait_asynccnt0();
        __syncthreads();

        v16bf af[4], bf[4];
#pragma unroll
        for (int i = 0; i < 4; ++i) {
            // A-frag (16x32): lane row = base + lmod; K = {8*kh..+8} U {16+8*kh..+8}
            const __bf16* p = &ldsA[(waveM * 64 + i * 16 + lmod) * GEMM_LDA + khalf * 8];
            af[i] = cat16(*(const v8bf*)p, *(const v8bf*)(p + 16));
        }
#pragma unroll
        for (int j = 0; j < 4; ++j) {
            // B-frag (32x16): lane col = base + lmod; K = {8*kh..+8} U {16+8*kh..+8}
            const __bf16* p = &ldsB[(waveN * 64 + j * 16 + lmod) * GEMM_LDB + khalf * 8];
            bf[j] = cat16(*(const v8bf*)p, *(const v8bf*)(p + 16));
        }
#pragma unroll
        for (int i = 0; i < 4; ++i)
#pragma unroll
            for (int j = 0; j < 4; ++j)
                acc[i][j] = wmma_bf16f32(af[i], bf[j], acc[i][j]);
    }

    // Epilogue: C/D layout -> VGPR r holds M = r + 8*khalf, N = lmod
    const int mbase = blockIdx.y * 128 + waveM * 64;
    const int nbase = blockIdx.x * 128 + waveN * 64;
#pragma unroll
    for (int i = 0; i < 4; ++i)
#pragma unroll
        for (int j = 0; j < 4; ++j)
#pragma unroll
            for (int r = 0; r < 8; ++r) {
                int row = mbase + i * 16 + r + 8 * khalf;
                int col = nbase + j * 16 + lmod;
                float val = acc[i][j][r];
                if (HAS_BIAS) val += bias[col];
                if (OUT_BF16)
                    ((__bf16*)Cout)[(size_t)row * Nn + col] = f2bf(val);
                else
                    ((float*)Cout)[(size_t)row * Nn + col] = val;
            }
}

// ---------------------------------------------------------------------------
// Flash attention: grid (N/64, H, B), 128 threads (4 waves, 16 q-rows each).
// Q,K packed in qk [B*N, 2C] (bf16), V [B*N, C] (bf16). Out fp32 [B*N, C].
// K tile staged via async global->LDS; V tile transposed through VGPRs.
// ---------------------------------------------------------------------------
#define FA_LDK 72   // 64 + 8 pad
#define FA_LDP 72

__global__ __launch_bounds__(128) void flash_attn_kernel(
    const __bf16* __restrict__ qk, const __bf16* __restrict__ v,
    float* __restrict__ out, int N, int C)
{
    const int HD   = 64;
    const int twoC = 2 * C;
    const int qt   = blockIdx.x;
    const int h    = blockIdx.y;
    const int b    = blockIdx.z;

    const int tid   = threadIdx.x;
    const int lane  = tid & 31;
    const int lmod  = lane & 15;
    const int khalf = lane >> 4;
    const int wave  = tid >> 5;

    const __bf16* qbase = qk + (size_t)b * N * twoC + h * HD;
    const __bf16* kbase = qk + (size_t)b * N * twoC + C + h * HD;
    const __bf16* vbase = v  + (size_t)b * N * C + h * HD;

    __shared__ __bf16 ldsK[64 * FA_LDK];
    __shared__ __bf16 ldsVt[64 * FA_LDK];
    __shared__ __bf16 ldsP[4][16 * FA_LDP];

    // Q A-fragments, loaded once: wave owns q rows [qt*64 + wave*16, +16)
    const int qrow = qt * 64 + wave * 16 + lmod;
    v16bf afq[2];
#pragma unroll
    for (int s = 0; s < 2; ++s) {
        const __bf16* p = qbase + (size_t)qrow * twoC + s * 32 + khalf * 8;
        afq[s] = cat16(*(const v8bf*)p, *(const v8bf*)(p + 16));
    }

    float m_i[8], l_i[8];
#pragma unroll
    for (int r = 0; r < 8; ++r) { m_i[r] = -1e30f; l_i[r] = 0.f; }
    v8f o[4] = {};

    const float scale = 0.125f;  // 1/sqrt(64)
    const int key = tid >> 1, hh = tid & 1;

    for (int kt = 0; kt < N / 64; ++kt) {
        __syncthreads();
        { // K tile: async DMA straight into LDS (no transpose needed)
            const __bf16* ks = kbase + (size_t)(kt * 64 + key) * twoC + hh * 32;
            __bf16* kd = &ldsK[key * FA_LDK + hh * 32];
#pragma unroll
            for (int c = 0; c < 4; ++c)
                async_ld_b128(kd + c * 8, ks + c * 8);
            // V tile: transpose into ldsVt[dim][key] via VGPRs
            const __bf16* vs = vbase + (size_t)(kt * 64 + key) * C + hh * 32;
#pragma unroll
            for (int c0 = 0; c0 < 32; c0 += 8) {
                v8bf vv = *(const v8bf*)(vs + c0);
#pragma unroll
                for (int e = 0; e < 8; ++e)
                    ldsVt[(hh * 32 + c0 + e) * FA_LDK + key] = vv[e];
            }
        }
        wait_asynccnt0();
        __syncthreads();

        // S = Q K^T   (16 q-rows x 64 keys per wave)
        v8f sfr[4] = {};
#pragma unroll
        for (int j = 0; j < 4; ++j)
#pragma unroll
            for (int ks = 0; ks < 2; ++ks) {
                const __bf16* p = &ldsK[(j * 16 + lmod) * FA_LDK + ks * 32 + khalf * 8];
                v16bf bf = cat16(*(const v8bf*)p, *(const v8bf*)(p + 16));
                sfr[j] = wmma_bf16f32(afq[ks], bf, sfr[j]);
            }

        // online softmax: row M = r + 8*khalf lives across the 16-lane group;
        // reductions are DPP row_xmask butterflies (VALU-only).
        float alpha[8];
#pragma unroll
        for (int r = 0; r < 8; ++r) {
            float mx = -1e30f;
#pragma unroll
            for (int j = 0; j < 4; ++j) mx = fmaxf(mx, sfr[j][r]);
            mx = rowmax16(mx) * scale;
            float mnew = fmaxf(m_i[r], mx);
            alpha[r] = __expf(m_i[r] - mnew);
            m_i[r] = mnew;
            float rs = 0.f;
#pragma unroll
            for (int j = 0; j < 4; ++j) {
                float p = __expf(sfr[j][r] * scale - mnew);
                sfr[j][r] = p;
                rs += p;
            }
            rs = rowsum16(rs);
            l_i[r] = l_i[r] * alpha[r] + rs;
        }
#pragma unroll
        for (int j = 0; j < 4; ++j)
#pragma unroll
            for (int r = 0; r < 8; ++r) o[j][r] *= alpha[r];

        // P: C-layout -> LDS -> A-layout (per-wave private region, bf16)
        __bf16* pw = &ldsP[wave][0];
#pragma unroll
        for (int j = 0; j < 4; ++j)
#pragma unroll
            for (int r = 0; r < 8; ++r)
                pw[(r + 8 * khalf) * FA_LDP + j * 16 + lmod] = f2bf(sfr[j][r]);

        // O += P * V
#pragma unroll
        for (int ks = 0; ks < 2; ++ks) {
            const __bf16* ap = pw + lmod * FA_LDP + ks * 32 + khalf * 8;
            v16bf af = cat16(*(const v8bf*)ap, *(const v8bf*)(ap + 16));
#pragma unroll
            for (int d = 0; d < 4; ++d) {
                const __bf16* bp = &ldsVt[(d * 16 + lmod) * FA_LDK + ks * 32 + khalf * 8];
                v16bf bf = cat16(*(const v8bf*)bp, *(const v8bf*)(bp + 16));
                o[d] = wmma_bf16f32(af, bf, o[d]);
            }
        }
    }

    // epilogue: out[b, n, h*64 + d] = O / l
#pragma unroll
    for (int r = 0; r < 8; ++r) {
        float invl = 1.0f / l_i[r];
        int rowg = qt * 64 + wave * 16 + r + 8 * khalf;
        float* orow = out + ((size_t)b * N + rowg) * C + h * HD;
#pragma unroll
        for (int d = 0; d < 4; ++d)
            orow[d * 16 + lmod] = o[d][r] * invl;
    }
}

// ---------------------------------------------------------------------------
// Launcher
// ---------------------------------------------------------------------------
extern "C" void kernel_launch(void* const* d_in, const int* in_sizes, int n_in,
                              void* d_out, int out_size, void* d_ws, size_t ws_size,
                              hipStream_t stream)
{
    (void)in_sizes; (void)n_in; (void)out_size; (void)ws_size;
    const float* x      = (const float*)d_in[0];
    const float* ln1g   = (const float*)d_in[1];
    const float* ln1b   = (const float*)d_in[2];
    const float* w_qk   = (const float*)d_in[3];
    const float* w_v    = (const float*)d_in[4];
    const float* ln2g   = (const float*)d_in[5];
    const float* ln2b   = (const float*)d_in[6];
    const float* w_proj = (const float*)d_in[7];
    const float* b_proj = (const float*)d_in[8];

    const int Bn = 2, N = 2048, C = 1024, H = 16;
    const int M = Bn * N;  // 4096

    char* ws = (char*)d_ws;
    size_t off = 0;
    auto carve = [&](size_t bytes) -> void* {
        void* p = ws + off;
        off += (bytes + 255) & ~(size_t)255;
        return p;
    };
    __bf16* xn     = (__bf16*)carve((size_t)M * C * 2);            // 8  MiB
    __bf16* wqkb   = (__bf16*)carve((size_t)C * 2 * C * 2);        // 4  MiB
    __bf16* wvb    = (__bf16*)carve((size_t)C * C * 2);            // 2  MiB
    __bf16* wpb    = (__bf16*)carve((size_t)C * C * 2);            // 2  MiB
    __bf16* qkb    = (__bf16*)carve((size_t)M * 2 * C * 2);        // 16 MiB
    __bf16* vb     = (__bf16*)carve((size_t)M * C * 2);            // 8  MiB
    float*  attn   = (float*)carve((size_t)M * C * 4);             // 16 MiB
    __bf16* ln2o   = (__bf16*)carve((size_t)M * C * 2);            // 8  MiB

    // 1) weight conversion fp32 -> bf16
    cvt_bf16_kernel<<<512, 256, 0, stream>>>(w_qk,   wqkb, C * 2 * C);
    cvt_bf16_kernel<<<512, 256, 0, stream>>>(w_v,    wvb,  C * C);
    cvt_bf16_kernel<<<512, 256, 0, stream>>>(w_proj, wpb,  C * C);

    // 2) LN1 (fp32 -> bf16)
    ln_bf16_kernel<<<M, 128, 0, stream>>>(x, ln1g, ln1b, xn, C);

    // 3) QK & V projections (bf16 WMMA, bf16 out)
    gemm_bf16_kernel<true, false><<<dim3((2 * C) / 128, M / 128), 128, 0, stream>>>(
        xn, wqkb, nullptr, qkb, M, 2 * C, C);
    gemm_bf16_kernel<true, false><<<dim3(C / 128, M / 128), 128, 0, stream>>>(
        xn, wvb, nullptr, vb, M, C, C);

    // 4) flash attention (WMMA S and PV, online softmax), fp32 out
    flash_attn_kernel<<<dim3(N / 64, H, Bn), 128, 0, stream>>>(qkb, vb, attn, N, C);

    // 5) LN2 (fp32 -> bf16)
    ln_bf16_kernel<<<M, 128, 0, stream>>>(attn, ln2g, ln2b, ln2o, C);

    // 6) output projection + bias (bf16 WMMA, fp32 out)
    gemm_bf16_kernel<false, true><<<dim3(C / 128, M / 128), 128, 0, stream>>>(
        ln2o, wpb, b_proj, (float*)d_out, M, C, C);
}
